// AlternatingHGN_50010599195033
// MI455X (gfx1250) — matrix operand order repaired and checked by reading, not compiled, not run on hardware
//
#include <hip/hip_runtime.h>
#include <cstdint>

#define NE_  100000
#define NNZ_ 500000

typedef __attribute__((ext_vector_type(16))) _Float16 v16h;
typedef __attribute__((ext_vector_type(2)))  _Float16 h2;
typedef __attribute__((ext_vector_type(8)))  float    v8f;

// ---------------------------------------------------------------------------
// CDNA5 async global->LDS staging (ASYNCcnt path, CDNA5 ISA)
// GVS mode: mem = SADDR(64b sgpr pair) + VADDR(32b vgpr offset) + IOFFSET.
// ---------------------------------------------------------------------------
__device__ __forceinline__ void stage_to_lds_async(float* lds_dst, const float* src,
                                                   int nfloats, int tid, int nthreads) {
  uint32_t base = (uint32_t)(uintptr_t)lds_dst;   // low 32b of flat LDS ptr == LDS offset
  int nchunks = nfloats >> 2;                      // 16B per lane per op
  for (int i = tid; i < nchunks; i += nthreads) {
    uint32_t loff = base + (uint32_t)(i << 4);
    int      goff = i << 4;
    asm volatile("global_load_async_to_lds_b128 %0, %1, %2"
                 :: "v"(loff), "v"(goff), "s"(src) : "memory");
  }
  asm volatile("s_wait_asynccnt 0" ::: "memory");
}

// ---------------------------------------------------------------------------
// f32 -> packed f16 helpers; A-tile fills are branch-free:
//   4x global float4 loads -> 8x v_cvt pack -> 2x ds_store_b128
// ---------------------------------------------------------------------------
__device__ __forceinline__ uint32_t pack2(float a, float b) {
  h2 p; p[0] = (_Float16)a; p[1] = (_Float16)b;
  return __builtin_bit_cast(uint32_t, p);
}

__device__ __forceinline__ void fill_a16_scaled(uint32_t* dst, const float* src, float sc) {
  const float4* s4 = (const float4*)src;
  float4 x0 = s4[0], x1 = s4[1], x2 = s4[2], x3 = s4[3];
  uint4 u0, u1;
  u0.x = pack2(x0.x * sc, x0.y * sc); u0.y = pack2(x0.z * sc, x0.w * sc);
  u0.z = pack2(x1.x * sc, x1.y * sc); u0.w = pack2(x1.z * sc, x1.w * sc);
  u1.x = pack2(x2.x * sc, x2.y * sc); u1.y = pack2(x2.z * sc, x2.w * sc);
  u1.z = pack2(x3.x * sc, x3.y * sc); u1.w = pack2(x3.z * sc, x3.w * sc);
  ((uint4*)dst)[0] = u0; ((uint4*)dst)[1] = u1;
}

__device__ __forceinline__ void fill_a16_bn(uint32_t* dst, const float* src,
                                            const float* mu, const float* isd) {
  const float4* s4 = (const float4*)src;
  const float4* m4 = (const float4*)mu;
  const float4* i4 = (const float4*)isd;
  uint4 u0, u1;
  {
    float4 x = s4[0], m = m4[0], i = i4[0];
    u0.x = pack2((x.x - m.x) * i.x, (x.y - m.y) * i.y);
    u0.y = pack2((x.z - m.z) * i.z, (x.w - m.w) * i.w);
  }
  {
    float4 x = s4[1], m = m4[1], i = i4[1];
    u0.z = pack2((x.x - m.x) * i.x, (x.y - m.y) * i.y);
    u0.w = pack2((x.z - m.z) * i.z, (x.w - m.w) * i.w);
  }
  {
    float4 x = s4[2], m = m4[2], i = i4[2];
    u1.x = pack2((x.x - m.x) * i.x, (x.y - m.y) * i.y);
    u1.y = pack2((x.z - m.z) * i.z, (x.w - m.w) * i.w);
  }
  {
    float4 x = s4[3], m = m4[3], i = i4[3];
    u1.z = pack2((x.x - m.x) * i.x, (x.y - m.y) * i.y);
    u1.w = pack2((x.z - m.z) * i.z, (x.w - m.w) * i.w);
  }
  ((uint4*)dst)[0] = u0; ((uint4*)dst)[1] = u1;
}

__device__ __forceinline__ void zero_a16(uint32_t* dst) {
  uint4 z; z.x = 0u; z.y = 0u; z.z = 0u; z.w = 0u;
  ((uint4*)dst)[0] = z; ((uint4*)dst)[1] = z;
}

// ---------------------------------------------------------------------------
// WMMA fragment loaders (wave32; A/B use contiguous k order, C/D as in ISA 7.12.2)
// A 16x32 f16: lane L: M=L&15; elem e: K = (L<16?0:16) + e
// B 32x16 f16: lane L: N=L&15; elem e: K = (L<16?0:16) + e
// C/D 16x16 f32: lane L: N=L&15; vgpr v: M = v + (L<16?0:8)
// ---------------------------------------------------------------------------
__device__ __forceinline__ v16h load_a_frag(const _Float16* a, int lda, int lane) {
  int m  = lane & 15;
  int kb = (lane < 16) ? 0 : 16;
  v16h f;
#pragma unroll
  for (int e = 0; e < 16; ++e) {
    int k = kb + e;
    f[e] = a[m * lda + k];
  }
  return f;
}

__device__ __forceinline__ v16h load_b_frag_f32(const float* b, int ldb, int n0, int lane) {
  int n  = n0 + (lane & 15);
  int kb = (lane < 16) ? 0 : 16;
  v16h f;
#pragma unroll
  for (int e = 0; e < 16; ++e) f[e] = (_Float16)b[(kb + e) * ldb + n];
  return f;
}

__device__ __forceinline__ void emit_tile_relu_stats(v8f acc, int n0, int mbase, int lane,
                                                     float* __restrict__ Z,
                                                     float* __restrict__ st, int e) {
  int n    = n0 + (lane & 15);
  int moff = (lane < 16) ? 0 : 8;
  float s = 0.f, s2 = 0.f;
#pragma unroll
  for (int v = 0; v < 8; ++v) {
    int m = mbase + moff + v;
    float x = acc[v];
    x = x > 0.f ? x : 0.f;
    if (m < NE_) { Z[(size_t)m * 64 + n] = x; s += x; s2 += x * x; }
  }
  atomicAdd(&st[e * 128 + n], s);
  atomicAdd(&st[e * 128 + 64 + n], s2);
}

__device__ __forceinline__ void store_tile(v8f acc, int n0, int mbase, int lane,
                                           float* __restrict__ D, int ldd) {
  int n    = n0 + (lane & 15);
  int moff = (lane < 16) ? 0 : 8;
#pragma unroll
  for (int v = 0; v < 8; ++v) {
    int m = mbase + moff + v;
    if (m < NE_) D[(size_t)m * ldd + n] = acc[v];
  }
}

// ---------------------------------------------------------------------------
// Phase 0: per-(relation,side) segment counts, then inv = 1/max(cnt,1)
// ---------------------------------------------------------------------------
__global__ void count_kernel(const int* __restrict__ r0, const int* __restrict__ c0,
                             const int* __restrict__ r1, const int* __restrict__ c1,
                             const int* __restrict__ r2, const int* __restrict__ c2,
                             float* __restrict__ cnt) {
  int t = blockIdx.x * blockDim.x + threadIdx.x;
  if (t >= 3 * NNZ_) return;
  int rel = t / NNZ_, n = t - rel * NNZ_;
  const int* rp = (rel == 0) ? r0 : (rel == 1) ? r1 : r2;
  const int* cp = (rel == 0) ? c0 : (rel == 1) ? c1 : c2;
  atomicAdd(&cnt[(size_t)(2 * rel) * NE_ + rp[n]], 1.0f);
  atomicAdd(&cnt[(size_t)(2 * rel + 1) * NE_ + cp[n]], 1.0f);
}

__global__ void inv_kernel(float* __restrict__ cnt) {
  int t = blockIdx.x * blockDim.x + threadIdx.x;
  if (t < 6 * NE_) cnt[t] = 1.0f / fmaxf(cnt[t], 1.0f);
}

// ---------------------------------------------------------------------------
// Scatter-add nnz values into per-entity sum buffers (segment sums)
// ---------------------------------------------------------------------------
__global__ void scatter_kernel(const float* __restrict__ data, const int* __restrict__ rows,
                               const int* __restrict__ cols, float* __restrict__ sumRow,
                               float* __restrict__ sumCol, int C) {
  int t = blockIdx.x * blockDim.x + threadIdx.x;
  if (t >= NNZ_ * C) return;
  int n = t / C, c = t - n * C;
  float v = data[t];
  atomicAdd(&sumRow[(size_t)rows[n] * C + c], v);
  if (sumCol) atomicAdd(&sumCol[(size_t)cols[n] * C + c], v);
}

// ---------------------------------------------------------------------------
// Pool GEMM: z[e] = relu( sum_{2 pairs} (sum_p * inv_p) @ pw[p] ), + BN stats.
// grid: (rowTiles, 3 entities), block 128 (4 waves x 16 rows x 64 cols).
// ---------------------------------------------------------------------------
template <int C>
__global__ __launch_bounds__(128)
void pool_wmma_kernel(const float* __restrict__ sums, const float* __restrict__ inv,
                      const float* __restrict__ pw, float* __restrict__ z,
                      float* __restrict__ stats) {
  constexpr int KSTEPS = (C + 31) >> 5;
  __shared__ __align__(16) float    wsh[2][KSTEPS * 32][64];
  __shared__ __align__(16) _Float16 ash[4][16][40];
  int tid = threadIdx.x;
  int e   = blockIdx.y;
  int p0  = (e == 0) ? 0 : (e == 1) ? 1 : 3;
  int p1  = (e == 0) ? 4 : (e == 1) ? 2 : 5;

  stage_to_lds_async(&wsh[0][0][0], pw + (size_t)p0 * C * 64, C * 64, tid, 128);
  stage_to_lds_async(&wsh[1][0][0], pw + (size_t)p1 * C * 64, C * 64, tid, 128);
  if (C < KSTEPS * 32) {  // zero-pad B rows K in [C, 32*KSTEPS)
    for (int idx = C * 64 + tid; idx < KSTEPS * 32 * 64; idx += 128) {
      (&wsh[0][0][0])[idx] = 0.0f;
      (&wsh[1][0][0])[idx] = 0.0f;
    }
  }
  __syncthreads();

  int wave  = tid >> 5, lane = tid & 31;
  int mbase = blockIdx.x * 64 + wave * 16;
  int mfill = lane >> 1;
  int k0    = (lane & 1) << 4;
  int rowf  = mbase + mfill; if (rowf > NE_ - 1) rowf = NE_ - 1;
  _Float16* A    = &ash[wave][0][0];
  uint32_t* Adst = (uint32_t*)&ash[wave][mfill][k0];

  v8f acc0 = {}, acc1 = {}, acc2 = {}, acc3 = {};
#pragma unroll
  for (int pi = 0; pi < 2; ++pi) {
    int p = pi ? p1 : p0;
    const float* Srow = sums + (size_t)p * NE_ * C + (size_t)rowf * C;
    float sc          = inv[(size_t)p * NE_ + rowf];
    const float* Bw   = &wsh[pi][0][0];
#pragma unroll
    for (int ks = 0; ks < KSTEPS; ++ks) {
      if (C >= 32 || k0 == 0)
        fill_a16_scaled(Adst, Srow + (ks << 5) + k0, sc);
      else
        zero_a16(Adst);
      v16h af = load_a_frag(A, 40, lane);
      v16h b0 = load_b_frag_f32(Bw + (ks << 5) * 64, 64, 0, lane);
      acc0 = __builtin_amdgcn_wmma_f32_16x16x32_f16(false, af, false, b0, (short)0, acc0, false, false);
      v16h b1 = load_b_frag_f32(Bw + (ks << 5) * 64, 64, 16, lane);
      acc1 = __builtin_amdgcn_wmma_f32_16x16x32_f16(false, af, false, b1, (short)0, acc1, false, false);
      v16h b2 = load_b_frag_f32(Bw + (ks << 5) * 64, 64, 32, lane);
      acc2 = __builtin_amdgcn_wmma_f32_16x16x32_f16(false, af, false, b2, (short)0, acc2, false, false);
      v16h b3 = load_b_frag_f32(Bw + (ks << 5) * 64, 64, 48, lane);
      acc3 = __builtin_amdgcn_wmma_f32_16x16x32_f16(false, af, false, b3, (short)0, acc3, false, false);
    }
  }
  float* Z = z + (size_t)e * NE_ * 64;
  emit_tile_relu_stats(acc0,  0, mbase, lane, Z, stats, e);
  emit_tile_relu_stats(acc1, 16, mbase, lane, Z, stats, e);
  emit_tile_relu_stats(acc2, 32, mbase, lane, Z, stats, e);
  emit_tile_relu_stats(acc3, 48, mbase, lane, Z, stats, e);
}

__global__ void bn_finalize_kernel(float* __restrict__ stats) {
  int t = threadIdx.x;                 // 192 = 3 entities x 64 cols
  if (t >= 192) return;
  int e = t / 64, n = t - e * 64;
  float s  = stats[e * 128 + n];
  float s2 = stats[e * 128 + 64 + n];
  float mu  = s / (float)NE_;
  float var = s2 / (float)NE_ - mu * mu;
  stats[384 + e * 128 + n]      = mu;
  stats[384 + e * 128 + 64 + n] = rsqrtf(var + 1e-5f);
}

// ---------------------------------------------------------------------------
// Table GEMM: T[p] = ((z[e]-mu)*invstd) @ bw[p].  grid: (rowTiles, 6 pairs).
// ---------------------------------------------------------------------------
__global__ __launch_bounds__(128)
void table_wmma_kernel(const float* __restrict__ z, const float* __restrict__ stats,
                       const float* __restrict__ bw, float* __restrict__ T) {
  __shared__ __align__(16) float    wsh[64][64];
  __shared__ __align__(16) _Float16 ash[4][16][40];
  __shared__ __align__(16) float    mush[64];
  __shared__ __align__(16) float    isdh[64];
  int tid = threadIdx.x;
  int p   = blockIdx.y;
  int e   = (p == 0 || p == 4) ? 0 : (p == 1 || p == 2) ? 1 : 2;

  stage_to_lds_async(&wsh[0][0], bw + (size_t)p * 64 * 64, 64 * 64, tid, 128);
  if (tid < 64) {
    mush[tid] = stats[384 + e * 128 + tid];
    isdh[tid] = stats[384 + e * 128 + 64 + tid];
  }
  __syncthreads();

  int wave  = tid >> 5, lane = tid & 31;
  int mbase = blockIdx.x * 64 + wave * 16;
  int mfill = lane >> 1;
  int k0    = (lane & 1) << 4;
  int rowf  = mbase + mfill; if (rowf > NE_ - 1) rowf = NE_ - 1;
  const float* Zrow = z + (size_t)e * NE_ * 64 + (size_t)rowf * 64;
  _Float16* A    = &ash[wave][0][0];
  uint32_t* Adst = (uint32_t*)&ash[wave][mfill][k0];

  v8f acc0 = {}, acc1 = {}, acc2 = {}, acc3 = {};
#pragma unroll
  for (int ks = 0; ks < 2; ++ks) {
    int kb = (ks << 5) + k0;
    fill_a16_bn(Adst, Zrow + kb, &mush[kb], &isdh[kb]);
    v16h af = load_a_frag(A, 40, lane);
    v16h b0 = load_b_frag_f32(&wsh[ks << 5][0], 64, 0, lane);
    acc0 = __builtin_amdgcn_wmma_f32_16x16x32_f16(false, af, false, b0, (short)0, acc0, false, false);
    v16h b1 = load_b_frag_f32(&wsh[ks << 5][0], 64, 16, lane);
    acc1 = __builtin_amdgcn_wmma_f32_16x16x32_f16(false, af, false, b1, (short)0, acc1, false, false);
    v16h b2 = load_b_frag_f32(&wsh[ks << 5][0], 64, 32, lane);
    acc2 = __builtin_amdgcn_wmma_f32_16x16x32_f16(false, af, false, b2, (short)0, acc2, false, false);
    v16h b3 = load_b_frag_f32(&wsh[ks << 5][0], 64, 48, lane);
    acc3 = __builtin_amdgcn_wmma_f32_16x16x32_f16(false, af, false, b3, (short)0, acc3, false, false);
  }
  float* Tp = T + (size_t)p * NE_ * 64;
  store_tile(acc0,  0, mbase, lane, Tp, 64);
  store_tile(acc1, 16, mbase, lane, Tp, 64);
  store_tile(acc2, 32, mbase, lane, Tp, 64);
  store_tile(acc3, 48, mbase, lane, Tp, 64);
}

// ---------------------------------------------------------------------------
// Broadcast: data[r][n] = relu(T[r,row][rows[n]] + T[r,col][cols[n]]); 16 thr/nnz.
// ---------------------------------------------------------------------------
__global__ void bcast_kernel(const float* __restrict__ T0, const float* __restrict__ T1,
                             const int* __restrict__ rows, const int* __restrict__ cols,
                             float* __restrict__ dataOut) {
  int t = blockIdx.x * blockDim.x + threadIdx.x;
  int n = t >> 4, seg = t & 15;
  if (n >= NNZ_) return;
  int r = rows[n], c = cols[n];
  float4 a = ((const float4*)(T0 + (size_t)r * 64))[seg];
  float4 b = ((const float4*)(T1 + (size_t)c * 64))[seg];
  float4 o;
  o.x = fmaxf(a.x + b.x, 0.f); o.y = fmaxf(a.y + b.y, 0.f);
  o.z = fmaxf(a.z + b.z, 0.f); o.w = fmaxf(a.w + b.w, 0.f);
  ((float4*)(dataOut + (size_t)n * 64))[seg] = o;
}

// ---------------------------------------------------------------------------
// Final: out = mean_p0 @ pw2[0,0] + mean_p4 @ pw2[2,0]  -> [NE, 16]
// ---------------------------------------------------------------------------
__global__ __launch_bounds__(128)
void final_wmma_kernel(const float* __restrict__ sums, const float* __restrict__ inv,
                       const float* __restrict__ pw2, float* __restrict__ out) {
  __shared__ __align__(16) float    wsh[2][64][16];
  __shared__ __align__(16) _Float16 ash[4][16][40];
  int tid = threadIdx.x;
  stage_to_lds_async(&wsh[0][0][0], pw2 + (size_t)0 * 64 * 16, 64 * 16, tid, 128);
  stage_to_lds_async(&wsh[1][0][0], pw2 + (size_t)4 * 64 * 16, 64 * 16, tid, 128);
  __syncthreads();

  int wave  = tid >> 5, lane = tid & 31;
  int mbase = blockIdx.x * 64 + wave * 16;
  int mfill = lane >> 1;
  int k0    = (lane & 1) << 4;
  int rowf  = mbase + mfill; if (rowf > NE_ - 1) rowf = NE_ - 1;
  _Float16* A    = &ash[wave][0][0];
  uint32_t* Adst = (uint32_t*)&ash[wave][mfill][k0];

  v8f acc = {};
#pragma unroll
  for (int pi = 0; pi < 2; ++pi) {
    int p = pi ? 4 : 0;
    const float* Srow = sums + (size_t)p * NE_ * 64 + (size_t)rowf * 64;
    float sc = inv[(size_t)p * NE_ + rowf];
#pragma unroll
    for (int ks = 0; ks < 2; ++ks) {
      fill_a16_scaled(Adst, Srow + (ks << 5) + k0, sc);
      v16h af = load_a_frag(A, 40, lane);
      v16h bf = load_b_frag_f32(&wsh[pi][ks << 5][0], 16, 0, lane);
      acc = __builtin_amdgcn_wmma_f32_16x16x32_f16(false, af, false, bf, (short)0, acc, false, false);
    }
  }
  int n    = lane & 15;
  int moff = (lane < 16) ? 0 : 8;
#pragma unroll
  for (int v = 0; v < 8; ++v) {
    int m = mbase + moff + v;
    if (m < NE_) out[(size_t)m * 16 + n] = acc[v];
  }
}

// ---------------------------------------------------------------------------
extern "C" void kernel_launch(void* const* d_in, const int* in_sizes, int n_in,
                              void* d_out, int out_size, void* d_ws, size_t ws_size,
                              hipStream_t stream) {
  (void)in_sizes; (void)n_in; (void)out_size; (void)ws_size;
  const float* vals[3] = {(const float*)d_in[0], (const float*)d_in[3], (const float*)d_in[6]};
  const int*   rows[3] = {(const int*)d_in[1], (const int*)d_in[4], (const int*)d_in[7]};
  const int*   cols[3] = {(const int*)d_in[2], (const int*)d_in[5], (const int*)d_in[8]};
  const float* pw0 = (const float*)d_in[9];
  const float* pw1 = (const float*)d_in[10];
  const float* pw2 = (const float*)d_in[11];
  const float* bw0 = (const float*)d_in[12];
  const float* bw1 = (const float*)d_in[13];
  float* out = (float*)d_out;

  // Workspace layout (floats): inv[6*NE] | sums[6*NE*64] | z[3*NE*64] |
  // stats[1024] | T[6*NE*64] | data[3*NNZ*64]   (~770 MB total)
  float* ws    = (float*)d_ws;
  float* invb  = ws;
  float* sums  = invb + (size_t)6 * NE_;
  float* zb    = sums + (size_t)6 * NE_ * 64;
  float* stats = zb + (size_t)3 * NE_ * 64;
  float* Tb    = stats + 1024;
  float* datab = Tb + (size_t)6 * NE_ * 64;

  const dim3 blk(256);
  const int  rowTiles = (NE_ + 63) / 64;

  // ---- phase 0: segment counts -> inverse means ----
  (void)hipMemsetAsync(invb, 0, (size_t)6 * NE_ * sizeof(float), stream);
  count_kernel<<<dim3((3 * NNZ_ + 255) / 256), blk, 0, stream>>>(
      rows[0], cols[0], rows[1], cols[1], rows[2], cols[2], invb);
  inv_kernel<<<dim3((6 * NE_ + 255) / 256), blk, 0, stream>>>(invb);

  // ---- layer 1 (C = 16) ----
  (void)hipMemsetAsync(sums, 0, (size_t)6 * NE_ * 16 * sizeof(float), stream);
  for (int r = 0; r < 3; ++r)
    scatter_kernel<<<dim3((NNZ_ * 16 + 255) / 256), blk, 0, stream>>>(
        vals[r], rows[r], cols[r],
        sums + (size_t)(2 * r) * NE_ * 16, sums + (size_t)(2 * r + 1) * NE_ * 16, 16);
  (void)hipMemsetAsync(stats, 0, 1024 * sizeof(float), stream);
  pool_wmma_kernel<16><<<dim3(rowTiles, 3), dim3(128), 0, stream>>>(sums, invb, pw0, zb, stats);
  bn_finalize_kernel<<<dim3(1), dim3(192), 0, stream>>>(stats);
  table_wmma_kernel<<<dim3(rowTiles, 6), dim3(128), 0, stream>>>(zb, stats, bw0, Tb);
  for (int r = 0; r < 3; ++r)
    bcast_kernel<<<dim3((NNZ_ * 16 + 255) / 256), blk, 0, stream>>>(
        Tb + (size_t)(2 * r) * NE_ * 64, Tb + (size_t)(2 * r + 1) * NE_ * 64,
        rows[r], cols[r], datab + (size_t)r * NNZ_ * 64);

  // ---- layer 2 (C = 64) ----
  (void)hipMemsetAsync(sums, 0, (size_t)6 * NE_ * 64 * sizeof(float), stream);
  for (int r = 0; r < 3; ++r)
    scatter_kernel<<<dim3((NNZ_ * 64 + 255) / 256), blk, 0, stream>>>(
        datab + (size_t)r * NNZ_ * 64, rows[r], cols[r],
        sums + (size_t)(2 * r) * NE_ * 64, sums + (size_t)(2 * r + 1) * NE_ * 64, 64);
  (void)hipMemsetAsync(stats, 0, 1024 * sizeof(float), stream);
  pool_wmma_kernel<64><<<dim3(rowTiles, 3), dim3(128), 0, stream>>>(sums, invb, pw1, zb, stats);
  bn_finalize_kernel<<<dim3(1), dim3(192), 0, stream>>>(stats);
  table_wmma_kernel<<<dim3(rowTiles, 6), dim3(128), 0, stream>>>(zb, stats, bw1, Tb);
  for (int r = 0; r < 3; ++r)
    bcast_kernel<<<dim3((NNZ_ * 16 + 255) / 256), blk, 0, stream>>>(
        Tb + (size_t)(2 * r) * NE_ * 64, Tb + (size_t)(2 * r + 1) * NE_ * 64,
        rows[r], cols[r], datab + (size_t)r * NNZ_ * 64);

  // ---- final pool: entity 0 only (pairs 0 and 4, row side) ----
  (void)hipMemsetAsync(sums, 0, (size_t)NE_ * 64 * sizeof(float), stream);
  (void)hipMemsetAsync(sums + (size_t)4 * NE_ * 64, 0, (size_t)NE_ * 64 * sizeof(float), stream);
  scatter_kernel<<<dim3((NNZ_ * 64 + 255) / 256), blk, 0, stream>>>(
      datab, rows[0], nullptr, sums, nullptr, 64);
  scatter_kernel<<<dim3((NNZ_ * 64 + 255) / 256), blk, 0, stream>>>(
      datab + (size_t)2 * NNZ_ * 64, rows[2], nullptr, sums + (size_t)4 * NE_ * 64, nullptr, 64);
  final_wmma_kernel<<<dim3(rowTiles), dim3(128), 0, stream>>>(sums, invb, pw2, out);
}
